// ViT_74809740362181
// MI455X (gfx1250) — matrix operand order repaired and checked by reading, not compiled, not run on hardware
//
#include <hip/hip_runtime.h>
#include <hip/hip_bf16.h>
#include <math.h>

// ---------------------------------------------------------------------------
// ViT-Base forward (B=32, S=197, D=768, NH=12, DH=64, L=12, MLP=3072, OUT=1000)
// All GEMMs via v_wmma_f32_16x16x32_bf16 (bf16 inputs, f32 accumulate).
// BM=128, BN=64, BK=64; 256 threads = 8 wave32 in 4x2 grid; each wave owns a
// 32x32 register tile (4 accumulators) -> 4 WMMA per K-half per wave.
// ---------------------------------------------------------------------------

typedef __attribute__((ext_vector_type(16))) __bf16 v16bf;
typedef __attribute__((ext_vector_type(8)))  __bf16 v8bf;
typedef __attribute__((ext_vector_type(4)))  __bf16 v4bf;
typedef __attribute__((ext_vector_type(8)))  float  v8f;

#define VIT_B   32
#define VIT_C   3
#define VIT_HW  224
#define VIT_NP  14
#define VIT_PS  16
#define VIT_D   768
#define VIT_NH  12
#define VIT_DH  64
#define VIT_L   12
#define VIT_S   197
#define VIT_SP  208      // padded leading dim for score rows (16-aligned)
#define VIT_MLP 3072
#define VIT_OUT 1000

#define EPI_NONE   0
#define EPI_BIAS   1
#define EPI_GELU   2
#define EPI_RESADD 3
#define EPI_PATCH  4

__device__ __forceinline__ float pos_val(int i, int j) {
    float ang = (float)i / powf(10000.0f, (float)j / (float)VIT_D);
    return (j & 1) ? cosf(ang) : sinf(ang);
}

// ---------------------------------------------------------------------------
// Generic batched WMMA GEMM: C[z] = A[z] (MxK) * B[z] (KxN)  (+ epilogue)
// z = zb*NH + zh with per-operand (b,h) strides.
// ---------------------------------------------------------------------------
template<int EPI, bool TRANSB>
__global__ __launch_bounds__(256)
void gemm_wmma(const float* __restrict__ A,  int lda, long sAb, long sAh,
               const float* __restrict__ Bm, int ldb, long sBb, long sBh,
               const float* __restrict__ bias, long sBiasH,
               float* __restrict__ Cc, int ldc, long sCb, long sCh,
               const float* __restrict__ resid,
               int M, int N, int K)
{
    __shared__ __bf16 As[128][64];   // [BM][BK] row-major           (16 KB)
    __shared__ __bf16 BsT[64][64];   // [BN][BK]  BsT[n][k] = B[k][n] (8 KB)

    const int z  = blockIdx.z;
    const int zb = z / VIT_NH;
    const int zh = z % VIT_NH;
    A  += zb * sAb + zh * sAh;
    Bm += zb * sBb + zh * sBh;
    if (bias) bias += zh * sBiasH;
    Cc += zb * sCb + zh * sCh;
    if (EPI == EPI_RESADD) resid += zb * sCb + zh * sCh;

    const int blockM = blockIdx.y * 128;
    const int blockN = blockIdx.x * 64;

    const int tid  = threadIdx.x;
    const int lane = tid & 31;
    const int wave = tid >> 5;
    const int wm   = wave >> 1;     // 0..3  (M sub-block of 32)
    const int wn   = wave & 1;      // 0..1  (N sub-block of 32)
    const int hi   = lane >> 4;
    const int l15  = lane & 15;

    v8f acc[2][2] = {};

    const int ktiles = (K + 63) >> 6;
    for (int kt = 0; kt < ktiles; ++kt) {
        const int k0 = kt << 6;

        // ---------------- stage A tile (128 x 64) ----------------
        if (blockM + 128 <= M && k0 + 64 <= K) {
#pragma unroll
            for (int i = 0; i < 8; ++i) {
                int q  = tid + i * 256;
                int r  = q >> 4, c4 = (q & 15) * 4;
                const float* ap = A + (long)(blockM + r) * lda + k0 + c4;
                float4 v = *(const float4*)ap;
                if (i == 0 && kt + 1 < ktiles)
                    __builtin_prefetch((const void*)(ap + 64), 0, 0);
                v4bf p = { (__bf16)v.x, (__bf16)v.y, (__bf16)v.z, (__bf16)v.w };
                *(v4bf*)&As[r][c4] = p;
            }
        } else {
            // clamped addresses + zero-select (branchless)
#pragma unroll
            for (int i = 0; i < 32; ++i) {
                int e = tid + i * 256;
                int r = e >> 6, c = e & 63;
                int gr = blockM + r, gk = k0 + c;
                int cr = gr < M ? gr : M - 1;
                int ck = gk < K ? gk : K - 1;
                float v = A[(long)cr * lda + ck];
                if (gr >= M || gk >= K) v = 0.f;
                As[r][c] = (__bf16)v;
            }
        }

        // ---------------- stage B tile (64k x 64n) -> BsT[n][k] ----------------
        if (k0 + 64 <= K && blockN + 64 <= N) {
            if (TRANSB) {
                // global is [n][k]: contiguous along k -> packed b128 stores
#pragma unroll
                for (int i = 0; i < 2; ++i) {
                    int q  = tid + i * 256;
                    int n  = q >> 3, k8 = (q & 7) * 8;
                    const float* bp = Bm + (long)(blockN + n) * ldb + k0 + k8;
                    float4 v0 = *(const float4*)bp;
                    float4 v1 = *(const float4*)(bp + 4);
                    if (i == 0 && kt + 1 < ktiles)
                        __builtin_prefetch((const void*)(bp + 64), 0, 0);
                    v8bf p = { (__bf16)v0.x, (__bf16)v0.y, (__bf16)v0.z, (__bf16)v0.w,
                               (__bf16)v1.x, (__bf16)v1.y, (__bf16)v1.z, (__bf16)v1.w };
                    *(v8bf*)&BsT[n][k8] = p;
                }
            } else {
                // global is [k][n]: coalesced along n, scattered b16 stores
#pragma unroll
                for (int i = 0; i < 4; ++i) {
                    int q  = tid + i * 256;
                    int kr = q >> 4, c4 = (q & 15) * 4;
                    const float* bp = Bm + (long)(k0 + kr) * ldb + blockN + c4;
                    float4 v = *(const float4*)bp;
                    if (i == 0 && kt + 1 < ktiles)
                        __builtin_prefetch((const void*)(bp + 64L * ldb), 0, 0);
                    BsT[c4 + 0][kr] = (__bf16)v.x;
                    BsT[c4 + 1][kr] = (__bf16)v.y;
                    BsT[c4 + 2][kr] = (__bf16)v.z;
                    BsT[c4 + 3][kr] = (__bf16)v.w;
                }
            }
        } else {
#pragma unroll
            for (int i = 0; i < 16; ++i) {
                int e = tid + i * 256;
                int kr = e >> 6, c = e & 63;
                int gk = k0 + kr, gn = blockN + c;
                int ck = gk < K ? gk : K - 1;
                int cn = gn < N ? gn : N - 1;
                float v = TRANSB ? Bm[(long)cn * ldb + ck]
                                 : Bm[(long)ck * ldb + cn];
                if (gk >= K || gn >= N) v = 0.f;
                BsT[c][kr] = (__bf16)v;
            }
        }
        __syncthreads();

        // -------- 2 K-halves x (2x2 register tile) = 8 WMMA per wave/tile ------
#pragma unroll
        for (int kk = 0; kk < 64; kk += 32) {
            v16bf af[2], bfv[2];
#pragma unroll
            for (int mi = 0; mi < 2; ++mi) {
                const __bf16* ap = &As[wm * 32 + mi * 16 + l15][kk + hi * 8];
                v8bf a0 = *(const v8bf*)ap;
                v8bf a1 = *(const v8bf*)(ap + 16);
                af[mi] = __builtin_shufflevector(a0, a1,
                    0,1,2,3,4,5,6,7,8,9,10,11,12,13,14,15);
            }
#pragma unroll
            for (int ni = 0; ni < 2; ++ni) {
                const __bf16* bp = &BsT[wn * 32 + ni * 16 + l15][kk + hi * 8];
                v8bf b0 = *(const v8bf*)bp;
                v8bf b1 = *(const v8bf*)(bp + 16);
                bfv[ni] = __builtin_shufflevector(b0, b1,
                    0,1,2,3,4,5,6,7,8,9,10,11,12,13,14,15);
            }
#pragma unroll
            for (int mi = 0; mi < 2; ++mi)
#pragma unroll
                for (int ni = 0; ni < 2; ++ni)
                    acc[mi][ni] = __builtin_amdgcn_wmma_f32_16x16x32_bf16(
                        false, af[mi], false, bfv[ni], (short)0,
                        acc[mi][ni], false, false);
        }
        __syncthreads();
    }

    // ---------------- epilogue (C/D layout: lanes0-15 M=r, lanes16-31 M=r+8) ---
#pragma unroll
    for (int mi = 0; mi < 2; ++mi) {
#pragma unroll
        for (int ni = 0; ni < 2; ++ni) {
            int n = blockN + wn * 32 + ni * 16 + l15;
#pragma unroll
            for (int r = 0; r < 8; ++r) {
                int m = blockM + wm * 32 + mi * 16 + r + hi * 8;
                if (m < M && n < N) {
                    float v = acc[mi][ni][r];
                    if (EPI == EPI_BIAS || EPI == EPI_GELU ||
                        EPI == EPI_RESADD || EPI == EPI_PATCH)
                        v += bias[n];
                    if (EPI == EPI_GELU)
                        v = 0.5f * v * (1.0f + erff(v * 0.70710678118654752f));
                    long orow = m;
                    if (EPI == EPI_PATCH) {
                        int bimg = m / 196, p = m % 196;
                        orow = (long)bimg * VIT_S + 1 + p;
                        v += pos_val(1 + p, n);
                    }
                    if (EPI == EPI_RESADD)
                        v += resid[orow * (long)ldc + n];
                    Cc[orow * (long)ldc + n] = v;
                }
            }
        }
    }
}

// ---------------------------------------------------------------------------
// Patchify: images [B,3,224,224] -> Pmat [B*196, 768] (channel-first per patch)
// ---------------------------------------------------------------------------
__global__ __launch_bounds__(256)
void patchify_k(const float* __restrict__ img, float* __restrict__ P, int n)
{
    int t = blockIdx.x * 256 + threadIdx.x;
    if (t >= n) return;
    int col = t % VIT_D;
    int row = t / VIT_D;
    int b = row / 196, p = row % 196;
    int pr = p / VIT_NP, pc = p % VIT_NP;
    int c  = col / 256;
    int rem = col % 256;
    int ii = rem / VIT_PS, jj = rem % VIT_PS;
    P[t] = img[(((long)b * VIT_C + c) * VIT_HW + (pr * VIT_PS + ii)) * VIT_HW
               + pc * VIT_PS + jj];
}

// x[b,0,:] = cls + pos(0,:)
__global__ __launch_bounds__(768)
void cls_k(const float* __restrict__ cls, float* __restrict__ x)
{
    int d = threadIdx.x;
    int b = blockIdx.x;
    x[(long)b * VIT_S * VIT_D + d] = cls[d] + pos_val(0, d);
}

// ---------------------------------------------------------------------------
// LayerNorm over D=768; one block (256 thr) per row; biased variance
// ---------------------------------------------------------------------------
__global__ __launch_bounds__(256)
void layernorm_k(const float* __restrict__ x, const float* __restrict__ g,
                 const float* __restrict__ bb, float* __restrict__ out)
{
    __shared__ float s1[256], s2[256];
    const long row = blockIdx.x;
    const float* xr = x + row * VIT_D;
    const int tid = threadIdx.x;
    float v0 = xr[tid], v1 = xr[tid + 256], v2 = xr[tid + 512];
    s1[tid] = v0 + v1 + v2;
    s2[tid] = v0 * v0 + v1 * v1 + v2 * v2;
    __syncthreads();
    for (int o = 128; o > 0; o >>= 1) {
        if (tid < o) { s1[tid] += s1[tid + o]; s2[tid] += s2[tid + o]; }
        __syncthreads();
    }
    float mean = s1[0] * (1.0f / VIT_D);
    float var  = s2[0] * (1.0f / VIT_D) - mean * mean;
    float rstd = rsqrtf(var + 1e-5f);
    float* orow = out + row * VIT_D;
    orow[tid]       = (v0 - mean) * rstd * g[tid]       + bb[tid];
    orow[tid + 256] = (v1 - mean) * rstd * g[tid + 256] + bb[tid + 256];
    orow[tid + 512] = (v2 - mean) * rstd * g[tid + 512] + bb[tid + 512];
}

// ---------------------------------------------------------------------------
// Softmax over rows of length 197 within padded ld=208 slices
// ---------------------------------------------------------------------------
__global__ __launch_bounds__(256)
void softmax_k(float* __restrict__ s)
{
    __shared__ float red[256];
    long z = blockIdx.x / VIT_S;
    int  qi = blockIdx.x % VIT_S;
    float* row = s + z * (long)(VIT_S * VIT_SP) + (long)qi * VIT_SP;
    const int tid = threadIdx.x;
    float v = (tid < VIT_S) ? row[tid] * 0.125f : -3.4e38f;
    red[tid] = v; __syncthreads();
    for (int o = 128; o > 0; o >>= 1) {
        if (tid < o) red[tid] = fmaxf(red[tid], red[tid + o]);
        __syncthreads();
    }
    float mx = red[0]; __syncthreads();
    float e = (tid < VIT_S) ? __expf(v - mx) : 0.f;
    red[tid] = e; __syncthreads();
    for (int o = 128; o > 0; o >>= 1) {
        if (tid < o) red[tid] += red[tid + o];
        __syncthreads();
    }
    float inv = 1.0f / red[0];
    if (tid < VIT_S) row[tid] = e * inv;
}

__global__ __launch_bounds__(256)
void add_k(float* __restrict__ x, const float* __restrict__ h, int n)
{
    int t = blockIdx.x * 256 + threadIdx.x;
    if (t < n) x[t] += h[t];
}

// ---------------------------------------------------------------------------
extern "C" void kernel_launch(void* const* d_in, const int* in_sizes, int n_in,
                              void* d_out, int out_size, void* d_ws, size_t ws_size,
                              hipStream_t stream)
{
    const float* images = (const float*)d_in[0];
    const float* W1     = (const float*)d_in[1];
    const float* b1     = (const float*)d_in[2];
    const float* cls    = (const float*)d_in[3];
    const float* ln1_g  = (const float*)d_in[4];
    const float* ln1_b  = (const float*)d_in[5];
    const float* Wq     = (const float*)d_in[6];
    const float* bq     = (const float*)d_in[7];
    const float* Wk     = (const float*)d_in[8];
    const float* bk     = (const float*)d_in[9];
    const float* Wv     = (const float*)d_in[10];
    const float* bv     = (const float*)d_in[11];
    const float* ln2_g  = (const float*)d_in[12];
    const float* ln2_b  = (const float*)d_in[13];
    const float* Wm1    = (const float*)d_in[14];
    const float* bm1    = (const float*)d_in[15];
    const float* Wm2    = (const float*)d_in[16];
    const float* bm2    = (const float*)d_in[17];
    const float* Wo     = (const float*)d_in[18];
    const float* bo     = (const float*)d_in[19];
    float* out = (float*)d_out;

    // workspace layout (floats)
    const long BSD = (long)VIT_B * VIT_S * VIT_D;             // 4,841,472
    float* X  = (float*)d_ws;                                  // [B,S,D]
    float* H  = X + BSD;                                       // scratch / attn out
    float* Q  = X + 2 * BSD;                                   // [B,NH,S,DH]
    float* Kt = X + 3 * BSD;
    float* V  = X + 4 * BSD;
    float* SC = X + 5 * BSD;                                   // [B,NH,S,SP]
    const long sSh = (long)VIT_S * VIT_SP;                     // 40976
    const long sSb = VIT_NH * sSh;
    float* MLPB = SC + (long)VIT_B * sSb;                      // [B,S,MLP]
    float* Pm = MLPB;                                          // patch matrix (reused)

    // ---- patch embed ----
    {
        int n = VIT_B * 196 * VIT_D;
        patchify_k<<<(n + 255) / 256, 256, 0, stream>>>(images, Pm, n);
        cls_k<<<VIT_B, VIT_D, 0, stream>>>(cls, X);
        // M = 6272 = 49*128 exact -> all-fast staging
        gemm_wmma<EPI_PATCH, false><<<dim3(12, 49, 1), 256, 0, stream>>>(
            Pm, VIT_D, 0, 0, W1, VIT_D, 0, 0, b1, 0,
            X, VIT_D, 0, 0, nullptr, VIT_B * 196, VIT_D, VIT_D);
    }

    const long sXb = (long)VIT_S * VIT_D;          // 151296 per image
    const long sQh = (long)VIT_S * VIT_DH;         // 12608 per head
    const long sQb = VIT_NH * sQh;                 // 151296 per image
    const int  rows = VIT_B * VIT_S;               // 6304
    const int  rY   = (rows + 127) / 128;          // 50

    for (int l = 0; l < VIT_L; ++l) {
        const float* g1 = ln1_g + (long)l * VIT_D;
        const float* be1 = ln1_b + (long)l * VIT_D;
        const float* g2 = ln2_g + (long)l * VIT_D;
        const float* be2 = ln2_b + (long)l * VIT_D;
        const float* wq = Wq + (long)l * VIT_NH * VIT_DH * VIT_DH;
        const float* wk = Wk + (long)l * VIT_NH * VIT_DH * VIT_DH;
        const float* wv = Wv + (long)l * VIT_NH * VIT_DH * VIT_DH;
        const float* bq_ = bq + (long)l * VIT_NH * VIT_DH;
        const float* bk_ = bk + (long)l * VIT_NH * VIT_DH;
        const float* bv_ = bv + (long)l * VIT_NH * VIT_DH;
        const float* wm1 = Wm1 + (long)l * VIT_D * VIT_MLP;
        const float* bm1_ = bm1 + (long)l * VIT_MLP;
        const float* wm2 = Wm2 + (long)l * VIT_MLP * VIT_D;
        const float* bm2_ = bm2 + (long)l * VIT_D;

        // h = LN1(x)
        layernorm_k<<<rows, 256, 0, stream>>>(X, g1, be1, H);

        // q/k/v: per (b,head)  h_slice[197x64] @ W[h][64x64] + b[h]
        dim3 gq(1, 2, VIT_B * VIT_NH);
        gemm_wmma<EPI_BIAS, false><<<gq, 256, 0, stream>>>(
            H, VIT_D, sXb, VIT_DH, wq, VIT_DH, 0, VIT_DH * VIT_DH, bq_, VIT_DH,
            Q, VIT_DH, sQb, sQh, nullptr, VIT_S, VIT_DH, VIT_DH);
        gemm_wmma<EPI_BIAS, false><<<gq, 256, 0, stream>>>(
            H, VIT_D, sXb, VIT_DH, wk, VIT_DH, 0, VIT_DH * VIT_DH, bk_, VIT_DH,
            Kt, VIT_DH, sQb, sQh, nullptr, VIT_S, VIT_DH, VIT_DH);
        gemm_wmma<EPI_BIAS, false><<<gq, 256, 0, stream>>>(
            H, VIT_D, sXb, VIT_DH, wv, VIT_DH, 0, VIT_DH * VIT_DH, bv_, VIT_DH,
            V, VIT_DH, sQb, sQh, nullptr, VIT_S, VIT_DH, VIT_DH);

        // scores = q @ k^T   [197 x 197] per (b,h), stored with ld=208
        gemm_wmma<EPI_NONE, true><<<dim3(4, 2, VIT_B * VIT_NH), 256, 0, stream>>>(
            Q, VIT_DH, sQb, sQh, Kt, VIT_DH, sQb, sQh, nullptr, 0,
            SC, VIT_SP, sSb, sSh, nullptr, VIT_S, VIT_S, VIT_DH);

        // softmax rows (scale 1/8 inside)
        softmax_k<<<VIT_B * VIT_NH * VIT_S, 256, 0, stream>>>(SC);

        // o = att @ v  -> written into H as [B,S,D] (col offset head*64)
        gemm_wmma<EPI_NONE, false><<<dim3(1, 2, VIT_B * VIT_NH), 256, 0, stream>>>(
            SC, VIT_SP, sSb, sSh, V, VIT_DH, sQb, sQh, nullptr, 0,
            H, VIT_D, sXb, VIT_DH, nullptr, VIT_S, VIT_DH, VIT_S);

        // x += o
        add_k<<<(int)((BSD + 255) / 256), 256, 0, stream>>>(X, H, (int)BSD);

        // h = LN2(x)
        layernorm_k<<<rows, 256, 0, stream>>>(X, g2, be2, H);

        // mlp = gelu(h @ Wm1 + bm1)
        gemm_wmma<EPI_GELU, false><<<dim3(48, rY, 1), 256, 0, stream>>>(
            H, VIT_D, 0, 0, wm1, VIT_MLP, 0, 0, bm1_, 0,
            MLPB, VIT_MLP, 0, 0, nullptr, rows, VIT_MLP, VIT_D);

        // x = x + (mlp @ Wm2 + bm2)
        gemm_wmma<EPI_RESADD, false><<<dim3(12, rY, 1), 256, 0, stream>>>(
            MLPB, VIT_MLP, 0, 0, wm2, VIT_D, 0, 0, bm2_, 0,
            X, VIT_D, 0, 0, X, rows, VIT_D, VIT_MLP);
    }

    // head: out = x[:,0] @ Wo + bo   (A row stride = S*D picks cls token)
    gemm_wmma<EPI_BIAS, false><<<dim3(16, 1, 1), 256, 0, stream>>>(
        X, (int)sXb, 0, 0, Wo, VIT_OUT, 0, 0, bo, 0,
        out, VIT_OUT, 0, 0, nullptr, VIT_B, VIT_OUT, VIT_D);
}
